// Mamba2_52209622450600
// MI455X (gfx1250) — matrix operand order repaired and checked by reading, not compiled, not run on hardware
//
#include <hip/hip_runtime.h>
#include <cstdint>
#include <cstddef>

#define SEQ      4096
#define DMODEL   1024
#define DINNER   2048
#define NHEADS   32
#define HEADDIM  64
#define DSTATE   128
#define DPROJ    4384
#define DCONVIN  2304
#define CHUNKSZ  128
#define NCHUNKS  32

typedef __bf16 bf16;
typedef __attribute__((ext_vector_type(16))) __bf16 v16bf;
typedef __attribute__((ext_vector_type(8)))  __bf16 v8bf;
typedef __attribute__((ext_vector_type(8)))  float  v8f;
typedef __attribute__((ext_vector_type(4)))  float  v4f;

__device__ inline v8f vzero8() {
  v8f z;
#pragma unroll
  for (int i = 0; i < 8; ++i) z[i] = 0.f;
  return z;
}

__device__ inline v8f wmma_bf16(v16bf a, v16bf b, v8f c) {
  // D = A(16x32 bf16) * B(32x16 bf16) + C(16x16 f32)
  return __builtin_amdgcn_wmma_f32_16x16x32_bf16(
      /*neg_a=*/false, a, /*neg_b=*/false, b,
      /*c_mod=*/(short)0, c, /*reuse_a=*/false, /*reuse_b=*/false);
}

// A-fragment (16x32, row-major source, K contiguous):
// lane<16 holds row M=lane, K = k0+0..15; lanes 16-31 hold K = k0+16..31.
__device__ inline v16bf load_fragA(const bf16* base, int ld, int row0, int k0, int lane) {
  const int r  = lane & 15;
  const int hs = (lane >> 4) & 1;
  const bf16* p = base + (size_t)(row0 + r) * ld + k0 + hs * 16;
  v8bf lo = *(const v8bf*)(p);
  v8bf hi = *(const v8bf*)(p + 8);
  v16bf f;
#pragma unroll
  for (int i = 0; i < 8; ++i) { f[i] = lo[i]; f[8 + i] = hi[i]; }
  return f;
}

// B-fragment (32x16): source is B^T row-major [N, K] (K contiguous).
// lane = column N (mod 16); lanes 0-15 hold K=k0+0..15, lanes 16-31 hold K=k0+16..31.
__device__ inline v16bf load_fragB(const bf16* base, int ld, int col0, int k0, int lane) {
  const bf16* p = base + (size_t)(col0 + (lane & 15)) * ld + k0 + ((lane >> 4) & 1) * 16;
  v8bf lo = *(const v8bf*)(p);
  v8bf hi = *(const v8bf*)(p + 8);
  v16bf f;
#pragma unroll
  for (int i = 0; i < 8; ++i) { f[i] = lo[i]; f[8 + i] = hi[i]; }
  return f;
}

// B-fragment sourced from f32 LDS (state matrix), converted on the fly.
__device__ inline v16bf load_fragB_f32(const float* base, int ld, int col0, int k0, int lane) {
  const float* p = base + (size_t)(col0 + (lane & 15)) * ld + k0 + ((lane >> 4) & 1) * 16;
  v16bf f;
#pragma unroll
  for (int q = 0; q < 4; ++q) {
    v4f x = *(const v4f*)(p + q * 4);
#pragma unroll
    for (int i = 0; i < 4; ++i) f[q * 4 + i] = (bf16)x[i];
  }
  return f;
}

// C/D layout: VGPR r, lanes 0-15: M=r, N=lane; lanes 16-31: M=8+r, N=lane-16.
__device__ inline void store_tile_f32(float* C, int ldc, int m0, int n0, int lane, v8f acc) {
  const int col  = n0 + (lane & 15);
  const int row0 = m0 + ((lane >> 4) & 1) * 8;
#pragma unroll
  for (int r = 0; r < 8; ++r) C[(size_t)(row0 + r) * ldc + col] = acc[r];
}

// 16-byte async global->LDS copy (gfx1250 ASYNCcnt path).
// lds_off = LDS byte address (low 32 bits of the flat __shared__ pointer).
__device__ inline void async_copy_b128(uint32_t lds_off, const void* gptr) {
  asm volatile("global_load_async_to_lds_b128 %0, %1, off"
               :: "v"(lds_off), "v"((uint64_t)(uintptr_t)gptr)
               : "memory");
}
__device__ inline void wait_async0() {
  asm volatile("s_wait_asynccnt 0x0" ::: "memory");
}

// ---------------------------------------------------------------------------
// Elementwise f32 -> bf16 conversion
// ---------------------------------------------------------------------------
__global__ __launch_bounds__(256) void cvt_f32_bf16(const float* __restrict__ in,
                                                    bf16* __restrict__ out, int n) {
  int i = blockIdx.x * 256 + threadIdx.x;
  if (i < n) out[i] = (bf16)in[i];
}

// ---------------------------------------------------------------------------
// NT GEMM: C[M,N] f32 = A[M,K] bf16 * B[N,K]^T bf16. Block = 8 waves covering
// 256 rows x 32 cols; each wave computes a 32x32 tile (2x2 WMMA tiles).
// ---------------------------------------------------------------------------
__global__ __launch_bounds__(256) void gemm_nt_bf16(const bf16* __restrict__ A,
                                                    const bf16* __restrict__ B,
                                                    float* __restrict__ C,
                                                    int K, int N) {
  const int lane = threadIdx.x & 31;
  const int w    = threadIdx.x >> 5;
  const int m0   = blockIdx.y * 256 + w * 32;
  const int n0   = blockIdx.x * 32;
  v8f a00 = vzero8(), a01 = vzero8(), a10 = vzero8(), a11 = vzero8();
  for (int k0 = 0; k0 < K; k0 += 32) {
    v16bf fa0 = load_fragA(A, K, m0,      k0, lane);
    v16bf fa1 = load_fragA(A, K, m0 + 16, k0, lane);
    v16bf fb0 = load_fragB(B, K, n0,      k0, lane);
    v16bf fb1 = load_fragB(B, K, n0 + 16, k0, lane);
    a00 = wmma_bf16(fa0, fb0, a00);
    a01 = wmma_bf16(fa0, fb1, a01);
    a10 = wmma_bf16(fa1, fb0, a10);
    a11 = wmma_bf16(fa1, fb1, a11);
  }
  store_tile_f32(C, N, m0,      n0,      lane, a00);
  store_tile_f32(C, N, m0,      n0 + 16, lane, a01);
  store_tile_f32(C, N, m0 + 16, n0,      lane, a10);
  store_tile_f32(C, N, m0 + 16, n0 + 16, lane, a11);
}

// ---------------------------------------------------------------------------
// Depthwise causal conv (d_conv=4) + SiLU over the xBC slice of zxbcdt,
// split into x (f32 + bf16), B (bf16), C (bf16).
// ---------------------------------------------------------------------------
__global__ __launch_bounds__(256) void conv_silu_split(
    const float* __restrict__ zx, const float* __restrict__ conv_w,
    const float* __restrict__ conv_b, float* __restrict__ x_f,
    bf16* __restrict__ x_bf, bf16* __restrict__ B_bf, bf16* __restrict__ C_bf) {
  int idx = blockIdx.x * 256 + threadIdx.x;
  if (idx >= SEQ * DCONVIN) return;
  int t = idx / DCONVIN;
  int c = idx - t * DCONVIN;
  float s = conv_b[c];
#pragma unroll
  for (int k = 0; k < 4; ++k) {
    int tt = t - 3 + k;
    if (tt >= 0) s += conv_w[c * 4 + k] * zx[(size_t)tt * DPROJ + DINNER + c];
  }
  float v = s / (1.f + expf(-s));   // SiLU
  if (c < DINNER) {
    size_t gi = (size_t)t * DINNER + c;
    x_f[gi]  = v;
    x_bf[gi] = (bf16)v;
  } else if (c < DINNER + DSTATE) {
    B_bf[(size_t)t * DSTATE + (c - DINNER)] = (bf16)v;
  } else {
    C_bf[(size_t)t * DSTATE + (c - DINNER - DSTATE)] = (bf16)v;
  }
}

// ---------------------------------------------------------------------------
// dt = softplus(raw + dt_bias); lg = dt * (-exp(A_log))  (log-decay per step)
// ---------------------------------------------------------------------------
__global__ __launch_bounds__(256) void dt_prep(const float* __restrict__ zx,
                                               const float* __restrict__ dt_bias,
                                               const float* __restrict__ A_log,
                                               float* __restrict__ dt_f,
                                               float* __restrict__ lg_f) {
  int idx = blockIdx.x * 256 + threadIdx.x;
  if (idx >= SEQ * NHEADS) return;
  int t = idx >> 5;
  int h = idx & 31;
  float v  = zx[(size_t)t * DPROJ + (2 * DINNER + 2 * DSTATE) + h] + dt_bias[h];
  float dt = (v > 20.f) ? v : log1pf(expf(v));
  float A  = -expf(A_log[h]);
  dt_f[idx] = dt;
  lg_f[idx] = dt * A;
}

// ---------------------------------------------------------------------------
// Chunked selective scan, attention form, all matmuls via WMMA bf16.
// One block (8 wave32s) per head; sequential over 32 chunks; state in LDS.
// Chunk C/B tiles are staged into LDS via the gfx1250 async-DMA path.
// ---------------------------------------------------------------------------
__global__ __launch_bounds__(256) void mamba_scan(
    const bf16* __restrict__ x_bf, const float* __restrict__ x_f,
    const bf16* __restrict__ B_bf, const bf16* __restrict__ C_bf,
    const float* __restrict__ dt_f, const float* __restrict__ lg_f,
    const float* __restrict__ Dvec, float* __restrict__ y_f) {
  constexpr int M_LD = 136, XT_LD = 136, XW_LD = 136, BT_LD = 136;
  constexpr int CB_LD = 136, ST_LD = 132;
  extern __shared__ char smem[];
  bf16*  sM  = (bf16*)smem;                   // [128][M_LD]  masked/weighted C.B^T
  bf16*  sXt = sM  + 128 * M_LD;              // [64][XT_LD]  X^T  ([p][j])
  bf16*  sXw = sXt + 64 * XT_LD;              // [64][XW_LD]  dtw-scaled X^T
  bf16*  sBt = sXw + 64 * XW_LD;              // [128][BT_LD] B^T  ([n][j])
  bf16*  sC  = sBt + 128 * BT_LD;             // [128][CB_LD] C tile (async-staged)
  bf16*  sB  = sC  + 128 * CB_LD;             // [128][CB_LD] B tile (async-staged)
  float* sSt = (float*)(sB + 128 * CB_LD);    // [64][ST_LD]  state [p][n] f32
  float* sLg = sSt + 64 * ST_LD;              // per-step log decay
  float* sLc = sLg + 128;                     // inclusive prefix of log decay
  float* sDt = sLc + 128;                     // dt
  float* sDw = sDt + 128;                     // dt * exp(lcum_last - lcum_j)
  float* sCd = sDw + 128;                     // cumdecay = exp(lcum_s)

  const int h    = blockIdx.x;
  const int tid  = threadIdx.x;
  const int lane = tid & 31;
  const int w    = tid >> 5;
  const int sb   = w * 16;                    // this wave's s-band
  const float Dh = Dvec[h];

  for (int i = tid; i < 64 * ST_LD; i += 256) sSt[i] = 0.f;
  __syncthreads();

  for (int c = 0; c < NCHUNKS; ++c) {
    const int c0 = c * CHUNKSZ;
    const bf16* Cg = C_bf + (size_t)c0 * DSTATE;  // shared across heads (ngroups=1)
    const bf16* Bg = B_bf + (size_t)c0 * DSTATE;

    // Kick off async staging of this chunk's C and B tiles (row-major copy,
    // 16B granules, padded LDS rows). Overlaps with the VALU staging below.
    for (int i = tid; i < 128 * 16; i += 256) {
      const int j = i >> 4, seg = (i & 15) * 8;
      async_copy_b128((uint32_t)(uintptr_t)(sC + j * CB_LD + seg),
                      Cg + (size_t)j * DSTATE + seg);
      async_copy_b128((uint32_t)(uintptr_t)(sB + j * CB_LD + seg),
                      Bg + (size_t)j * DSTATE + seg);
    }

    if (tid < 128) {
      sDt[tid] = dt_f[(size_t)(c0 + tid) * NHEADS + h];
      sLg[tid] = lg_f[(size_t)(c0 + tid) * NHEADS + h];
    }
    __syncthreads();
    if (tid == 0) {
      float run = 0.f;
      for (int s = 0; s < 128; ++s) { run += sLg[s]; sLc[s] = run; }
    }
    __syncthreads();
    if (tid < 128) {
      const float last = sLc[127];
      sCd[tid] = expf(sLc[tid]);
      sDw[tid] = sDt[tid] * expf(last - sLc[tid]);
    }
    __syncthreads();

    // Stage X^T (and dtw-scaled X^T) and B^T into LDS (transposed -> VALU path).
    for (int i = tid; i < 64 * 128; i += 256) {
      const int p = i & 63, j = i >> 6;
      const bf16 v = x_bf[(size_t)(c0 + j) * DINNER + h * HEADDIM + p];
      sXt[p * XT_LD + j] = v;
      sXw[p * XW_LD + j] = (bf16)((float)v * sDw[j]);
    }
    for (int i = tid; i < 128 * 128; i += 256) {
      const int n = i & 127, j = i >> 7;
      sBt[n * BT_LD + j] = Bg[(size_t)j * DSTATE + n];
    }
    wait_async0();        // C/B tiles resident in LDS
    __syncthreads();

    // G = C . B^T over n, then M[s,j] = G * dt_j * exp(lcum_s - lcum_j), j<=s.
    {
      const int jlane = lane & 15;
      const int sB0   = sb + ((lane >> 4) & 1) * 8;
      for (int jb = 0; jb < 8; ++jb) {
        v8f acc = vzero8();
#pragma unroll
        for (int k0 = 0; k0 < DSTATE; k0 += 32)
          acc = wmma_bf16(load_fragA(sC, CB_LD, sb, k0, lane),
                          load_fragB(sB, CB_LD, jb * 16, k0, lane), acc);
        const int jidx = jb * 16 + jlane;
        const float dtj = sDt[jidx], lcj = sLc[jidx];
#pragma unroll
        for (int r = 0; r < 8; ++r) {
          const int s = sB0 + r;
          const float m = (jidx <= s) ? acc[r] * dtj * expf(sLc[s] - lcj) : 0.f;
          sM[s * M_LD + jidx] = (bf16)m;
        }
      }
    }
    __syncthreads();

    // Y = M @ X  +  diag(cumdecay) * (C @ state^T)  +  D * x
    for (int pb = 0; pb < 4; ++pb) {
      v8f acc  = vzero8();
      v8f acc2 = vzero8();
#pragma unroll
      for (int k0 = 0; k0 < CHUNKSZ; k0 += 32)
        acc = wmma_bf16(load_fragA(sM, M_LD, sb, k0, lane),
                        load_fragB(sXt, XT_LD, pb * 16, k0, lane), acc);
#pragma unroll
      for (int k0 = 0; k0 < DSTATE; k0 += 32)
        acc2 = wmma_bf16(load_fragA(sC, CB_LD, sb, k0, lane),
                         load_fragB_f32(sSt, ST_LD, pb * 16, k0, lane), acc2);
      const int p  = pb * 16 + (lane & 15);
      const int s0 = sb + ((lane >> 4) & 1) * 8;
#pragma unroll
      for (int r = 0; r < 8; ++r) {
        const int s = s0 + r;
        const size_t gi = (size_t)(c0 + s) * DINNER + h * HEADDIM + p;
        y_f[gi] = acc[r] + sCd[s] * acc2[r] + Dh * x_f[gi];
      }
    }
    __syncthreads();

    // state' = cdl * state + (dtw .* X)^T @ B   (32 tiles: 4 p-tiles x 8 n-tiles)
    {
      const float cdl = sCd[127];
      for (int q = 0; q < 4; ++q) {
        const int id = w * 4 + q;
        const int pt = id >> 3, nt = id & 7;
        v8f acc = vzero8();
#pragma unroll
        for (int k0 = 0; k0 < CHUNKSZ; k0 += 32)
          acc = wmma_bf16(load_fragA(sXw, XW_LD, pt * 16, k0, lane),
                          load_fragB(sBt, BT_LD, nt * 16, k0, lane), acc);
        const int n  = nt * 16 + (lane & 15);
        const int p0 = pt * 16 + ((lane >> 4) & 1) * 8;
#pragma unroll
        for (int r = 0; r < 8; ++r) {
          const int p = p0 + r;
          const float old = sSt[p * ST_LD + n];
          sSt[p * ST_LD + n] = acc[r] + cdl * old;
        }
      }
    }
    __syncthreads();
  }
}

// ---------------------------------------------------------------------------
// RMSNorm over d_inner + SiLU(z) gate -> bf16 for the out-projection GEMM.
// ---------------------------------------------------------------------------
__global__ __launch_bounds__(256) void norm_gate(const float* __restrict__ y,
                                                 const float* __restrict__ zx,
                                                 const float* __restrict__ nw,
                                                 const float* __restrict__ nb,
                                                 bf16* __restrict__ out) {
  __shared__ float red[256];
  const int t = blockIdx.x;
  float ss = 0.f;
  for (int ccol = threadIdx.x; ccol < DINNER; ccol += 256) {
    const float v = y[(size_t)t * DINNER + ccol];
    ss += v * v;
  }
  red[threadIdx.x] = ss;
  __syncthreads();
  for (int off = 128; off > 0; off >>= 1) {
    if (threadIdx.x < off) red[threadIdx.x] += red[threadIdx.x + off];
    __syncthreads();
  }
  const float inv = rsqrtf(red[0] / (float)DINNER + 1e-6f);
  for (int ccol = threadIdx.x; ccol < DINNER; ccol += 256) {
    const size_t gi = (size_t)t * DINNER + ccol;
    const float v = y[gi] * inv * nw[ccol] + nb[ccol];
    const float z = zx[(size_t)t * DPROJ + ccol];
    out[gi] = (bf16)(v * (z / (1.f + expf(-z))));
  }
}

// ---------------------------------------------------------------------------
extern "C" void kernel_launch(void* const* d_in, const int* in_sizes, int n_in,
                              void* d_out, int out_size, void* d_ws, size_t ws_size,
                              hipStream_t stream) {
  (void)in_sizes; (void)n_in; (void)out_size; (void)ws_size;
  const float* u       = (const float*)d_in[0];
  const float* W_in    = (const float*)d_in[1];
  const float* conv_w  = (const float*)d_in[2];
  const float* conv_b  = (const float*)d_in[3];
  const float* W_out   = (const float*)d_in[4];
  const float* norm_w  = (const float*)d_in[5];
  const float* norm_b  = (const float*)d_in[6];
  const float* dt_bias = (const float*)d_in[7];
  const float* A_log   = (const float*)d_in[8];
  const float* Dvec    = (const float*)d_in[9];

  char* ws = (char*)d_ws;
  auto take = [&ws](size_t bytes) -> char* {
    char* p = ws;
    ws += (bytes + 255) & ~(size_t)255;
    return p;
  };
  bf16*  u_bf    = (bf16*)take((size_t)SEQ * DMODEL * sizeof(bf16));
  bf16*  Win_bf  = (bf16*)take((size_t)DPROJ * DMODEL * sizeof(bf16));
  bf16*  Wout_bf = (bf16*)take((size_t)DMODEL * DINNER * sizeof(bf16));
  float* zx      = (float*)take((size_t)SEQ * DPROJ * sizeof(float));
  float* x_f     = (float*)take((size_t)SEQ * DINNER * sizeof(float));
  bf16*  x_bf    = (bf16*)take((size_t)SEQ * DINNER * sizeof(bf16));
  bf16*  Bb      = (bf16*)take((size_t)SEQ * DSTATE * sizeof(bf16));
  bf16*  Cb      = (bf16*)take((size_t)SEQ * DSTATE * sizeof(bf16));
  float* dt_f    = (float*)take((size_t)SEQ * NHEADS * sizeof(float));
  float* lg_f    = (float*)take((size_t)SEQ * NHEADS * sizeof(float));
  float* y_f     = (float*)take((size_t)SEQ * DINNER * sizeof(float));
  bf16*  g_bf    = (bf16*)take((size_t)SEQ * DINNER * sizeof(bf16));

  // 1. bf16 conversions (u, W_in, W_out)
  {
    int n0 = SEQ * DMODEL;
    cvt_f32_bf16<<<dim3((n0 + 255) / 256), 256, 0, stream>>>(u, u_bf, n0);
    int n1 = DPROJ * DMODEL;
    cvt_f32_bf16<<<dim3((n1 + 255) / 256), 256, 0, stream>>>(W_in, Win_bf, n1);
    int n2 = DMODEL * DINNER;
    cvt_f32_bf16<<<dim3((n2 + 255) / 256), 256, 0, stream>>>(W_out, Wout_bf, n2);
  }

  // 2. in-projection: zxbcdt[4096,4384] = u_bf @ W_in^T  (WMMA bf16)
  gemm_nt_bf16<<<dim3(DPROJ / 32, SEQ / 256), 256, 0, stream>>>(
      u_bf, Win_bf, zx, DMODEL, DPROJ);

  // 3. causal depthwise conv + SiLU + split
  {
    int n = SEQ * DCONVIN;
    conv_silu_split<<<dim3((n + 255) / 256), 256, 0, stream>>>(
        zx, conv_w, conv_b, x_f, x_bf, Bb, Cb);
  }

  // 4. dt / log-decay preparation
  {
    int n = SEQ * NHEADS;
    dt_prep<<<dim3((n + 255) / 256), 256, 0, stream>>>(zx, dt_bias, A_log, dt_f, lg_f);
  }

  // 5. chunked scan (one block per head), WMMA + async-LDS staging
  {
    const size_t scan_smem =
        (size_t)(128 * 136 + 64 * 136 + 64 * 136 + 128 * 136 +
                 128 * 136 + 128 * 136) * sizeof(bf16) +
        (size_t)(64 * 132 + 5 * 128) * sizeof(float);   // 210,432 B < 320 KB
    mamba_scan<<<dim3(NHEADS), 256, scan_smem, stream>>>(
        x_bf, x_f, Bb, Cb, dt_f, lg_f, Dvec, y_f);
  }

  // 6. RMSNorm + SiLU(z) gate -> bf16
  norm_gate<<<dim3(SEQ), 256, 0, stream>>>(y_f, zx, norm_w, norm_b, g_bf);

  // 7. out-projection: out[4096,1024] = gated @ W_out^T (WMMA bf16, f32 out)
  gemm_nt_bf16<<<dim3(DMODEL / 32, SEQ / 256), 256, 0, stream>>>(
      g_bf, Wout_bf, (float*)d_out, DINNER, DMODEL);
}
